// SemanticGNN_56221121904753
// MI455X (gfx1250) — matrix operand order repaired and checked by reading, not compiled, not run on hardware
//
#include <hip/hip_runtime.h>
#include <hip/hip_bf16.h>

#define Bn  64
#define Nn  512
#define FIN 21
#define Hn  64
#define Ln  3
#define Cn  9

typedef __attribute__((ext_vector_type(4)))  float    v4f;
typedef __attribute__((ext_vector_type(8)))  float    v8f;
typedef __attribute__((ext_vector_type(4)))  _Float16 v4h;
typedef __attribute__((ext_vector_type(8)))  _Float16 v8h;
typedef __attribute__((ext_vector_type(16))) _Float16 v16h;
typedef __attribute__((ext_vector_type(4)))  int      v4i;

#if defined(__HIP_DEVICE_COMPILE__) && defined(__has_builtin)
#if __has_builtin(__builtin_amdgcn_global_load_async_to_lds_b128) && \
    __has_builtin(__builtin_amdgcn_s_wait_asynccnt)
#define USE_ASYNC_LDS 1
#endif
#endif

__device__ __forceinline__ v16h joinv(v8h lo, v8h hi){
  v16h r;
#pragma unroll
  for(int i=0;i<8;++i){ r[i]=lo[i]; r[8+i]=hi[i]; }
  return r;
}

// ---------- weight prep: WlT16[l][col][k] = f16(Wl[l][k][col]) ----------
__global__ void prep_wl_k(const float* __restrict__ Wl, _Float16* __restrict__ WlT16){
  for(int e = threadIdx.x + blockIdx.x*blockDim.x; e < Ln*Hn*Hn; e += blockDim.x*gridDim.x){
    int l = e/(Hn*Hn), r = e%(Hn*Hn);
    int k = r/Hn, c = r%Hn;
    WlT16[(l*Hn + c)*Hn + k] = (_Float16)Wl[(l*Hn + k)*Hn + c];
  }
}

// ---------- node embedding: h = relu(X @ We + be) ----------
__global__ void embed_k(const float* __restrict__ X, const float* __restrict__ We,
                        const float* __restrict__ be, float* __restrict__ h){
  int tid = threadIdx.x + blockIdx.x*blockDim.x;   // over B*N*H
  int row = tid >> 6;                              // b*N + n
  int c   = tid & 63;
  float acc = be[c];
#pragma unroll
  for(int f=0; f<FIN; ++f) acc += X[row*FIN + f] * We[f*Hn + c];
  h[tid] = fmaxf(acc, 0.f);
}

// ---------- hW = h @ Wl[l]  (WMMA f16 -> f32), output transposed f16 ----------
__global__ void __launch_bounds__(128) gemm_hw_k(const float* __restrict__ h,
            const _Float16* __restrict__ WlT16, int l, _Float16* __restrict__ hWT){
  __shared__ __attribute__((aligned(32))) _Float16 hs16[16*Hn];     // A rows as f16
  __shared__ __attribute__((aligned(32))) _Float16 out16c[Hn*16];   // C tile, col-major
  const int t = threadIdx.x;
  const int wv = t >> 5, lane = t & 31;
  const int i0 = blockIdx.x * 16;          // global row in [0, B*N)
  const int b  = i0 >> 9;
  const int n0 = i0 & 511;

  // 16 rows x 64 cols are contiguous in h: vector-convert 4 floats at a time
  for(int e = t; e < (16*Hn)/4; e += 128){
    v4f x = *(const v4f*)&h[(size_t)i0*Hn + e*4];
    v4h hx;
#pragma unroll
    for(int u=0;u<4;++u) hx[u]=(_Float16)x[u];
    *(v4h*)&hs16[e*4] = hx;
  }
  __syncthreads();

  const int c0   = wv*16;
  const int arow = lane & 15;
  const int col  = c0 + (lane & 15);
  v8f acc = {};
#pragma unroll
  for(int kt=0; kt<2; ++kt){
    // A 16x32 f16 fragment: lanes0-15 K=kb..kb+15; lanes16-31 +16 (contiguous, matches B)
    int kb = kt*32 + ((lane<16)?0:16);
    v8h alo = *(const v8h*)&hs16[arow*Hn + kb];
    v8h ahi = *(const v8h*)&hs16[arow*Hn + kb + 8];
    v16h afrag = joinv(alo, ahi);
    // B 32x16 f16 fragment: lane = column, 16 contiguous K halves
    int koff = kt*32 + ((lane<16)?0:16);
    v16h bfrag = *(const v16h*)&WlT16[(l*Hn + col)*Hn + koff];
    acc = __builtin_amdgcn_wmma_f32_16x16x32_f16(false, afrag, false, bfrag,
                                                 (short)0, acc, false, false);
  }
  { // C layout: VGPR r -> (M=r, N=lane) lanes0-15 ; (M=8+r, N=lane-16) lanes16-31
    int mo = (lane < 16) ? 0 : 8;
#pragma unroll
    for(int r=0;r<8;++r) out16c[col*16 + r + mo] = (_Float16)acc[r];
  }
  __syncthreads();
  if (t < Hn){ // coalesced 32B transposed store
    const float4* s4 = (const float4*)&out16c[t*16];
    float4* d4 = (float4*)&hWT[((size_t)(b*Hn + t))*Nn + n0];
    d4[0] = s4[0];
    d4[1] = s4[1];
  }
}

// ---------- src/dst attention scores (8 nodes per thread, v8h loads) ----------
__global__ void scores_k(const _Float16* __restrict__ hWT, const float* __restrict__ a_src,
                         const float* __restrict__ a_dst, int l,
                         float* __restrict__ sv, float* __restrict__ dv){
  int tid = threadIdx.x + blockIdx.x*blockDim.x;   // over B*N/8
  int b  = tid >> 6;
  int n0 = (tid & 63) * 8;
  v4f s0={}, s1={}, d0={}, d1={};
#pragma unroll
  for(int c=0;c<Hn;++c){
    v8h hv = *(const v8h*)&hWT[((size_t)(b*Hn + c))*Nn + n0];
    float as = a_src[l*Hn + c], ad = a_dst[l*Hn + c];
#pragma unroll
    for(int e=0;e<4;++e){
      float v0 = (float)hv[e], v1 = (float)hv[e+4];
      s0[e] += v0*as; s1[e] += v1*as;
      d0[e] += v0*ad; d1[e] += v1*ad;
    }
  }
  *(v4f*)&sv[b*Nn + n0]     = s0;
  *(v4f*)&sv[b*Nn + n0 + 4] = s1;
  *(v4f*)&dv[b*Nn + n0]     = d0;
  *(v4f*)&dv[b*Nn + n0 + 4] = d1;
}

// ---------- masked softmax attention + WMMA aggregate + residual + layernorm ----------
__global__ void __launch_bounds__(128) attn_k(const float* __restrict__ adj,
        const float* __restrict__ sv, const float* __restrict__ dv,
        const _Float16* __restrict__ hWT, const float* __restrict__ gamma,
        const float* __restrict__ beta, int l, float* __restrict__ h){
  __shared__ __attribute__((aligned(16))) float sc[16*Nn];          // 32 KB scores -> probs
  __shared__ __attribute__((aligned(32))) _Float16 p16[16*Nn];      // 16 KB f16 probs (A op)
  __shared__ __attribute__((aligned(16))) float sdst[Nn];
  __shared__ float ssrc[16];
  __shared__ __attribute__((aligned(16))) float outf[16*Hn];        // attn @ hW (f32)
  __shared__ float red[16*8], red2[16*8], mrow[16], vrow[16], mm[16], rr[16];
  const int t = threadIdx.x;
  const int wv = t>>5, lane = t&31;
  const int b  = blockIdx.y;
  const int i0 = blockIdx.x*16;
  const float NEG = -__builtin_inff();

  // stage dst scores into LDS: async DMA (ASYNCcnt) when available, 16B per lane
#if defined(USE_ASYNC_LDS)
  {
    v4i* gpv = (v4i*)(dv + (size_t)b*Nn);          // generic reinterpret (drops const)
    v4i* lpv = (v4i*)&sdst[0];
    __builtin_amdgcn_global_load_async_to_lds_b128(
        (__attribute__((address_space(1))) v4i*)(gpv + t),
        (__attribute__((address_space(3))) v4i*)(lpv + t),
        0, 0);
    __builtin_amdgcn_s_wait_asynccnt(0);
  }
#else
  for(int j=t;j<Nn;j+=128) sdst[j] = dv[b*Nn + j];
#endif
  if (t<16) ssrc[t] = sv[b*Nn + i0 + t];
  __syncthreads();

  const int i = t>>3, q = t&7;   // 8 threads per row; each owns 16 float4 chunks
  const v4f* arow4 = (const v4f*)&adj[((size_t)b*Nn + i0 + i)*Nn];
  const v4f* sdst4 = (const v4f*)sdst;
  const float sI = ssrc[i];
  float lm = NEG;
#pragma unroll 4
  for(int k=0;k<16;++k){
    int j4 = q + (k<<3);               // float4 index; j = 4*j4
    v4f a4 = arow4[j4];                // global_load_b128, 128B / 8-lane group
    v4f d4 = sdst4[j4];
    v4f v;
#pragma unroll
    for(int e=0;e<4;++e){
      float x = sI + d4[e];
      x = (x > 0.f) ? x : 0.2f*x;      // leaky_relu 0.2
      v[e] = (a4[e] == 0.f) ? NEG : x; // adjacency mask
      lm = fmaxf(lm, v[e]);
    }
    *(v4f*)&sc[i*Nn + j4*4] = v;
  }
  red[i*8+q] = lm;
  __syncthreads();
  if (t < 16){
    float m = red[t*8];
#pragma unroll
    for(int r=1;r<8;++r) m = fmaxf(m, red[t*8+r]);
    mrow[t] = m;
  }
  __syncthreads();
  {
    float m = mrow[i], ls = 0.f;
#pragma unroll 4
    for(int k=0;k<16;++k){
      int j4 = q + (k<<3);
      v4f v = *(const v4f*)&sc[i*Nn + j4*4];
      v4f p;
#pragma unroll
      for(int e=0;e<4;++e){
        p[e] = (v[e] == NEG) ? 0.f : __expf(v[e] - m);   // all-masked row -> 0
        ls += p[e];
      }
      *(v4f*)&sc[i*Nn + j4*4] = p;
    }
    red[i*8+q] = ls;
  }
  __syncthreads();
  if (t<16){
    float sum=0.f;
#pragma unroll
    for(int r=0;r<8;++r) sum += red[t*8+r];
    vrow[t] = (sum > 0.f) ? (1.f/sum) : 0.f;             // nan_to_num(softmax)
  }
  __syncthreads();
  {
    float inv = vrow[i];
#pragma unroll 4
    for(int k=0;k<16;++k){
      int j4 = q + (k<<3);
      v4f p = *(const v4f*)&sc[i*Nn + j4*4];
      v4h ph;
#pragma unroll
      for(int e=0;e<4;++e) ph[e] = (_Float16)(p[e]*inv);
      *(v4h*)&p16[i*Nn + j4*4] = ph;                     // ds_store_b64
    }
  }
  __syncthreads();

  // P[16x512] @ hW[512 x 16-col tile], 16 WMMA K-steps, f32 accumulate
  const int c0  = wv*16;
  const int arw = lane & 15;
  const int col = c0 + (lane & 15);
  const _Float16* bbase = &hWT[((size_t)(b*Hn + col))*Nn];
  v8f acc = {};
#pragma unroll
  for(int kt=0; kt<16; ++kt){
    int j0 = kt*32;
    int kb = j0 + ((lane<16)?0:16);
    v8h alo = *(const v8h*)&p16[arw*Nn + kb];
    v8h ahi = *(const v8h*)&p16[arw*Nn + kb + 8];
    v16h afrag = joinv(alo, ahi);
    int koff = j0 + ((lane<16)?0:16);
    v16h bfrag = *(const v16h*)&bbase[koff];
    acc = __builtin_amdgcn_wmma_f32_16x16x32_f16(false, afrag, false, bfrag,
                                                 (short)0, acc, false, false);
  }
  {
    int mo = (lane<16)?0:8;
#pragma unroll
    for(int r=0;r<8;++r) outf[(r+mo)*Hn + col] = acc[r];
  }
  __syncthreads();

  // residual + layernorm (f32 stats), h updated in place (block owns its rows)
  float* hp = &h[((size_t)b*Nn + i0 + i)*Hn + q*8];
  v4f xa = *(const v4f*)hp     + *(const v4f*)&outf[i*Hn + q*8];
  v4f xb = *(const v4f*)(hp+4) + *(const v4f*)&outf[i*Hn + q*8 + 4];
  float ps=0.f, ps2=0.f;
#pragma unroll
  for(int e=0;e<4;++e){ ps += xa[e]+xb[e]; ps2 += xa[e]*xa[e] + xb[e]*xb[e]; }
  red[i*8+q]=ps; red2[i*8+q]=ps2;
  __syncthreads();
  if (t<16){
    float s1=0.f, s2=0.f;
#pragma unroll
    for(int r=0;r<8;++r){ s1+=red[t*8+r]; s2+=red2[t*8+r]; }
    float mu = s1*(1.f/Hn);
    float var = s2*(1.f/Hn) - mu*mu;
    mm[t]=mu; rr[t]=rsqrtf(var + 1e-5f);
  }
  __syncthreads();
  {
    float mu = mm[i], rs = rr[i];
    v4f ga = *(const v4f*)&gamma[l*Hn + q*8];
    v4f gb = *(const v4f*)&gamma[l*Hn + q*8 + 4];
    v4f ba = *(const v4f*)&beta [l*Hn + q*8];
    v4f bb = *(const v4f*)&beta [l*Hn + q*8 + 4];
    v4f oa, ob;
#pragma unroll
    for(int e=0;e<4;++e){
      oa[e] = (xa[e]-mu)*rs*ga[e] + ba[e];
      ob[e] = (xb[e]-mu)*rs*gb[e] + bb[e];
    }
    *(v4f*)hp     = oa;
    *(v4f*)(hp+4) = ob;
  }
}

// ---------- attention pooling + classifier head (one block per batch) ----------
__global__ void __launch_bounds__(256) pool_k(const float* __restrict__ h,
      const unsigned char* __restrict__ mask,
      const float* __restrict__ P1, const float* __restrict__ pb1,
      const float* __restrict__ P2, const float* __restrict__ pb2,
      const float* __restrict__ C1, const float* __restrict__ cb1,
      const float* __restrict__ C2, const float* __restrict__ cb2,
      float* __restrict__ out){
  __shared__ float pa[Nn];
  __shared__ float red[256];
  __shared__ float gpart[4*Hn];
  __shared__ float glds[Hn];
  __shared__ float r1[Hn];
  const int t = threadIdx.x;
  const int b = blockIdx.x;
  const float NEG = -__builtin_inff();

#pragma unroll
  for(int half=0; half<2; ++half){
    int n = t + half*256;
    v4f hr[16];
    const v4f* hp4 = (const v4f*)&h[((size_t)b*Nn + n)*Hn];
#pragma unroll
    for(int kk=0;kk<16;++kk) hr[kk] = hp4[kk];
    float acc = pb2[0];
    for(int c=0;c<Hn;++c){
      float s0 = pb1[c];
#pragma unroll
      for(int k=0;k<Hn;++k) s0 += hr[k>>2][k&3]*P1[k*Hn + c];
      acc += tanhf(s0)*P2[c];
    }
    pa[n] = mask[b*Nn + n] ? acc : NEG;
  }
  __syncthreads();
  // softmax over 512 nodes
  red[t] = fmaxf(pa[t], pa[t+256]);
  __syncthreads();
  for(int s=128;s>0;s>>=1){ if(t<s) red[t]=fmaxf(red[t],red[t+s]); __syncthreads(); }
  float m = red[0];
  __syncthreads();
  float p0 = (pa[t]     == NEG) ? 0.f : __expf(pa[t]     - m);
  float p1 = (pa[t+256] == NEG) ? 0.f : __expf(pa[t+256] - m);
  pa[t]=p0; pa[t+256]=p1;
  red[t]=p0+p1;
  __syncthreads();
  for(int s=128;s>0;s>>=1){ if(t<s) red[t]+=red[t+s]; __syncthreads(); }
  float inv = (red[0] > 0.f) ? (1.f/red[0]) : 0.f;
  __syncthreads();
  pa[t]*=inv; pa[t+256]*=inv;
  __syncthreads();
  { // g[hc] = sum_n pa[n]*h[b,n,hc]
    int hc = t & 63, part = t >> 6;
    float acc=0.f;
    for(int n=part*128; n<part*128+128; ++n)
      acc += pa[n]*h[((size_t)b*Nn + n)*Hn + hc];
    gpart[part*Hn + hc]=acc;
  }
  __syncthreads();
  if (t<Hn) glds[t] = gpart[t]+gpart[Hn+t]+gpart[2*Hn+t]+gpart[3*Hn+t];
  __syncthreads();
  if (t<Hn){
    float s0 = cb1[t];
#pragma unroll
    for(int k=0;k<Hn;++k) s0 += glds[k]*C1[k*Hn+t];
    r1[t]=fmaxf(s0,0.f);
  }
  __syncthreads();
  if (t<Cn){
    float o = cb2[t];
#pragma unroll
    for(int k=0;k<Hn;++k) o += r1[k]*C2[k*Cn+t];
    out[b*Cn+t]=o;
  }
}

extern "C" void kernel_launch(void* const* d_in, const int* in_sizes, int n_in,
                              void* d_out, int out_size, void* d_ws, size_t ws_size,
                              hipStream_t stream){
  (void)in_sizes; (void)n_in; (void)out_size; (void)ws_size;
  const float* X     = (const float*)d_in[0];
  const float* adj   = (const float*)d_in[1];
  const unsigned char* mask = (const unsigned char*)d_in[2];
  const float* We    = (const float*)d_in[3];
  const float* be    = (const float*)d_in[4];
  const float* Wl    = (const float*)d_in[5];
  const float* a_src = (const float*)d_in[6];
  const float* a_dst = (const float*)d_in[7];
  const float* gamma = (const float*)d_in[8];
  const float* beta  = (const float*)d_in[9];
  const float* P1    = (const float*)d_in[10];
  const float* pb1   = (const float*)d_in[11];
  const float* P2    = (const float*)d_in[12];
  const float* pb2   = (const float*)d_in[13];
  const float* C1    = (const float*)d_in[14];
  const float* cb1   = (const float*)d_in[15];
  const float* C2    = (const float*)d_in[16];
  const float* cb2   = (const float*)d_in[17];
  float* out = (float*)d_out;

  // workspace carve-out (~12.9 MB)
  char* w = (char*)d_ws;
  float*    h     = (float*)w;    w += (size_t)Bn*Nn*Hn*sizeof(float);     // 8 MB
  _Float16* hWT   = (_Float16*)w; w += (size_t)Bn*Hn*Nn*sizeof(_Float16);  // 4 MB
  float*    sv    = (float*)w;    w += (size_t)Bn*Nn*sizeof(float);
  float*    dv    = (float*)w;    w += (size_t)Bn*Nn*sizeof(float);
  _Float16* WlT16 = (_Float16*)w; w += (size_t)Ln*Hn*Hn*sizeof(_Float16);

  prep_wl_k<<<12, 256, 0, stream>>>(Wl, WlT16);
  embed_k<<<(Bn*Nn*Hn)/256, 256, 0, stream>>>(X, We, be, h);
  for(int l=0;l<Ln;++l){
    gemm_hw_k<<<(Bn*Nn)/16, 128, 0, stream>>>(h, WlT16, l, hWT);
    scores_k<<<(Bn*Nn/8)/256, 256, 0, stream>>>(hWT, a_src, a_dst, l, sv, dv);
    attn_k<<<dim3(Nn/16, Bn), 128, 0, stream>>>(adj, sv, dv, hWT, gamma, beta, l, h);
  }
  pool_k<<<Bn, 256, 0, stream>>>(h, mask, P1, pb1, P2, pb2, C1, cb1, C2, cb2, out);
}
